// Encoder_26663156974095
// MI455X (gfx1250) — matrix operand order repaired and checked by reading, not compiled, not run on hardware
//
#include <hip/hip_runtime.h>
#include <hip/hip_bf16.h>
#include <stdint.h>

#define N_NODES  10000
#define M_PAD    10112           // padded row count for bf16 hidden (128-multiple)
#define E_EDGES  160000
#define EPRIME   170000          // E + N self loops
#define HEADS    4
#define OUT_COLS 228             // 128 + 100
#define NEG_SLOPE 0.2f

typedef __bf16 bf16_t;
typedef bf16_t bf16x8  __attribute__((ext_vector_type(8)));
typedef bf16_t bf16x16 __attribute__((ext_vector_type(16)));
typedef float  floatx4 __attribute__((ext_vector_type(4)));
typedef float  floatx8 __attribute__((ext_vector_type(8)));

union BF16X16 { bf16x16 v; bf16x8 h[2]; };

__device__ __forceinline__ void async_copy_b128(unsigned lds_off, const void* gptr) {
  // CDNA5 GLOBAL_LOAD_ASYNC_TO_LDS_B128: per-lane 16B global -> LDS, ASYNCcnt-tracked.
  asm volatile("global_load_async_to_lds_b128 %0, %1, off"
               :: "v"(lds_off), "v"((unsigned long long)(uintptr_t)gptr)
               : "memory");
}
__device__ __forceinline__ void wait_async0() {
  asm volatile("s_wait_asynccnt 0" ::: "memory");
}

// ---------------------------------------------------------------------------
// WMMA GEMM: C[M,N] = A[M,K] @ B[K,N]
//   MODE 0: A fp32 (async-staged raw f32; cvt at fragment build),
//           epilogue = relu(acc + bias[col]) stored bf16
//   MODE 1: A bf16 (async-staged raw), epilogue = acc stored fp32.
//           Caller guarantees A rows readable up to a 128-row multiple.
// Tiles: block = 256 threads (8 wave32), M_TILE=128, N_TILE=64, K_TILE=32.
// Each wave owns a 32x32 patch = 4x v_wmma_f32_16x16x32_bf16 accumulators.
// LDS strides padded (A: 36 f32 / 40 bf16, B: 40 bf16) so the 16-lane
// fragment reads hit 16 distinct bank-group starts (gcd(stride/4,64)=4).
// ---------------------------------------------------------------------------
template <int MODE>
__global__ __launch_bounds__(256) void gemm_wmma_kernel(
    const void* __restrict__ Av, const float* __restrict__ B,
    const float* __restrict__ bias, void* __restrict__ Outv,
    int M, int N, int K)
{
  constexpr int A_BYTES = (MODE == 0) ? 128 * 36 * 4 : 128 * 40 * 2;
  __shared__ __attribute__((aligned(16))) char  AsRaw[A_BYTES];
  __shared__ __attribute__((aligned(16))) bf16_t Bs[64 * 40];  // col-major, ld=40

  const int tid   = threadIdx.x;
  const int wave  = tid >> 5;
  const int lane  = tid & 31;
  const int l15   = lane & 15;
  const int lhalf = lane >> 4;
  const int wm    = (wave >> 1) * 32;   // 4 row groups
  const int wn    = (wave & 1) * 32;    // 2 col groups
  const int mtile = blockIdx.x * 128;
  const int ntile = blockIdx.y * 64;

  const unsigned asBase = (unsigned)(uintptr_t)(void*)AsRaw; // LDS byte offset

  floatx8 acc[2][2] = {};

  for (int kt = 0; kt < K; kt += 32) {
    __syncthreads();   // protect LDS of previous iteration

    // ---- A tile via async global->LDS copies (raw element type) ----
    if (MODE == 0) {
      const float* Af = (const float*)Av;
      for (int i = tid; i < 1024; i += 256) {        // 128 rows x 8 chunks(16B)
        int row = i >> 3, ch = i & 7;
        int grow = mtile + row;
        unsigned lds_off = asBase + row * 144 + ch * 16;
        if (grow < M)
          async_copy_b128(lds_off, Af + (size_t)grow * K + kt + ch * 4);
      }
    } else {
      const bf16_t* Ab = (const bf16_t*)Av;
      for (int i = tid; i < 512; i += 256) {         // 128 rows x 4 chunks(16B)
        int row = i >> 2, ch = i & 3;
        int grow = mtile + row;                      // always readable (padded)
        unsigned lds_off = asBase + row * 80 + ch * 16;
        async_copy_b128(lds_off, Ab + (size_t)grow * K + kt + ch * 8);
      }
    }

    // ---- B tile (32x64 fp32) -> LDS col-major bf16 ----
    for (int i = tid; i < 512; i += 256) {           // 512 groups of 4 elems
      int kr = i >> 4;
      int c4 = (i & 15) << 2;
      int gn = ntile + c4;
      const float* bp = B + (size_t)(kt + kr) * N + gn;
      floatx4 v = {0.f, 0.f, 0.f, 0.f};
      if (gn + 3 < N) {
        v = *(const floatx4*)bp;
      } else {
        for (int j = 0; j < 4; ++j) if (gn + j < N) v[j] = bp[j];
      }
      for (int j = 0; j < 4; ++j) Bs[(c4 + j) * 40 + kr] = (bf16_t)v[j];
    }

    wait_async0();
    __syncthreads();

    // ---- build fragments ----
    BF16X16 afrag[2], bfrag[2];
    if (MODE == 0) {
      const float* AsF = (const float*)AsRaw;
      for (int mg = 0; mg < 2; ++mg) {
        int row = wm + 16 * mg + l15;
        const float* ap = AsF + row * 36 + (lhalf ? 8 : 0);
        floatx4 f0 = *(const floatx4*)ap;            // K = k0..k0+3
        floatx4 f1 = *(const floatx4*)(ap + 4);      // K = k0+4..k0+7
        floatx4 f2 = *(const floatx4*)(ap + 16);     // K = k0+16..k0+19
        floatx4 f3 = *(const floatx4*)(ap + 20);     // K = k0+20..k0+23
        BF16X16 a;
        for (int j = 0; j < 4; ++j) {
          a.v[j]      = (bf16_t)f0[j];
          a.v[4 + j]  = (bf16_t)f1[j];
          a.v[8 + j]  = (bf16_t)f2[j];
          a.v[12 + j] = (bf16_t)f3[j];
        }
        afrag[mg] = a;
      }
    } else {
      const bf16_t* AsB = (const bf16_t*)AsRaw;
      for (int mg = 0; mg < 2; ++mg) {
        int row = wm + 16 * mg + l15;
        const bf16_t* ap = AsB + row * 40 + (lhalf ? 8 : 0);
        afrag[mg].h[0] = *(const bf16x8*)ap;
        afrag[mg].h[1] = *(const bf16x8*)(ap + 16);
      }
    }
    for (int ng = 0; ng < 2; ++ng) {
      int col = wn + 16 * ng + l15;
      const bf16_t* bp = &Bs[col * 40 + (lhalf ? 8 : 0)];
      bfrag[ng].h[0] = *(const bf16x8*)bp;
      bfrag[ng].h[1] = *(const bf16x8*)(bp + 16);
    }

    for (int mg = 0; mg < 2; ++mg)
      for (int ng = 0; ng < 2; ++ng)
        acc[mg][ng] = __builtin_amdgcn_wmma_f32_16x16x32_bf16(
            false, afrag[mg].v, false, bfrag[ng].v,
            (short)0, acc[mg][ng], false, false);
  }

  // ---- epilogue ----
  for (int mg = 0; mg < 2; ++mg)
    for (int ng = 0; ng < 2; ++ng)
      for (int r = 0; r < 8; ++r) {
        int grow = mtile + wm + 16 * mg + r + lhalf * 8;
        int gcol = ntile + wn + 16 * ng + l15;
        if (grow < M && gcol < N) {
          float v = acc[mg][ng][r];
          if (MODE == 0) {
            v += bias[gcol];
            v = v > 0.f ? v : 0.f;
            ((bf16_t*)Outv)[(size_t)grow * N + gcol] = (bf16_t)v;
          } else {
            ((float*)Outv)[(size_t)grow * N + gcol] = v;
          }
        }
      }
}

// ---------------------------------------------------------------------------
// Edge-phase kernels (one conv at a time; H=4 fixed)
// ---------------------------------------------------------------------------
__device__ __forceinline__ void atomicMaxF(float* addr, float v) {
  if (v >= 0.f) atomicMax((int*)addr, __float_as_int(v));
  else          atomicMin((unsigned int*)addr, __float_as_uint(v));
}

__global__ __launch_bounds__(256) void fill_u32_kernel(unsigned int* p, unsigned int v, long long n) {
  long long i = (long long)blockIdx.x * blockDim.x + threadIdx.x;
  if (i < n) p[i] = v;
}

// Pass 1: one wave per (edge, head): logit = att . leaky_relu(xl[src]+xr[dst]);
// store logit and atomic-max into per-(dst,head) running max.
__global__ __launch_bounds__(256) void edge_logits_max_kernel(
    const float* __restrict__ xl, const float* __restrict__ xr,
    const float* __restrict__ att, const int* __restrict__ edge_index,
    float* __restrict__ logits, float* __restrict__ mbuf, int C)
{
  long long gid = (long long)blockIdx.x * blockDim.x + threadIdx.x;
  int gwave = (int)(gid >> 5);
  int lane  = threadIdx.x & 31;
  if (gwave >= EPRIME * HEADS) return;
  int e = gwave >> 2;       // / HEADS
  int h = gwave & 3;
  int s, d;
  if (e < E_EDGES) { s = edge_index[e]; d = edge_index[E_EDGES + e]; }
  else             { s = e - E_EDGES;   d = s; }
  const int HC = HEADS * C;
  const float* pl = xl + (size_t)s * HC + h * C;
  const float* pr = xr + (size_t)d * HC + h * C;
  const float* pa = att + h * C;
  float acc = 0.f;
  for (int c = lane; c < C; c += 32) {
    float v = pl[c] + pr[c];
    v = v > 0.f ? v : NEG_SLOPE * v;
    acc += pa[c] * v;
  }
  for (int off = 16; off > 0; off >>= 1) acc += __shfl_xor(acc, off, 32);
  if (lane == 0) {
    logits[gwave] = acc;
    atomicMaxF(mbuf + d * HEADS + h, acc);
  }
}

// Pass 2: p = exp(logit - max); accumulate denom per (dst, head).
__global__ __launch_bounds__(256) void edge_softmax_kernel(
    const int* __restrict__ edge_index, float* __restrict__ logits,
    const float* __restrict__ mbuf, float* __restrict__ denom)
{
  long long idx = (long long)blockIdx.x * blockDim.x + threadIdx.x;
  if (idx >= (long long)EPRIME * HEADS) return;
  int e = (int)(idx >> 2);
  int h = (int)(idx & 3);
  int d = (e < E_EDGES) ? edge_index[E_EDGES + e] : (e - E_EDGES);
  float p = __expf(logits[idx] - mbuf[d * HEADS + h]);
  logits[idx] = p;
  atomicAdd(denom + d * HEADS + h, p);
}

// Pass 3: out[dst] += alpha * xl[src]  (wave per (edge,head), lane-strided C)
__global__ __launch_bounds__(256) void edge_scatter_kernel(
    const float* __restrict__ xl, const int* __restrict__ edge_index,
    const float* __restrict__ logits, const float* __restrict__ denom,
    float* __restrict__ accum, int C)
{
  long long gid = (long long)blockIdx.x * blockDim.x + threadIdx.x;
  int gwave = (int)(gid >> 5);
  int lane  = threadIdx.x & 31;
  if (gwave >= EPRIME * HEADS) return;
  int e = gwave >> 2;
  int h = gwave & 3;
  int s, d;
  if (e < E_EDGES) { s = edge_index[e]; d = edge_index[E_EDGES + e]; }
  else             { s = e - E_EDGES;   d = s; }
  const int HC = HEADS * C;
  float alpha = logits[gwave] / denom[d * HEADS + h];
  const float* pl = xl + (size_t)s * HC + h * C;
  float* pd = accum + (size_t)d * HC + h * C;
  for (int c = lane; c < C; c += 32) atomicAdd(pd + c, alpha * pl[c]);
}

// Head-mean + bias into the output slice.
__global__ __launch_bounds__(256) void finalize_kernel(
    const float* __restrict__ accum, const float* __restrict__ bias,
    float* __restrict__ obase, int C, int coloff)
{
  long long idx = (long long)blockIdx.x * blockDim.x + threadIdx.x;
  if (idx >= (long long)N_NODES * C) return;
  int n = (int)(idx / C);
  int c = (int)(idx % C);
  const int HC = HEADS * C;
  float s = 0.f;
  for (int h = 0; h < HEADS; ++h) s += accum[(size_t)n * HC + h * C + c];
  obase[(size_t)n * OUT_COLS + coloff + c] = 0.25f * s + bias[c];
}

// ---------------------------------------------------------------------------
static inline int cdiv_i(long long a, long long b) { return (int)((a + b - 1) / b); }

extern "C" void kernel_launch(void* const* d_in, const int* in_sizes, int n_in,
                              void* d_out, int out_size, void* d_ws, size_t ws_size,
                              hipStream_t stream)
{
  (void)in_sizes; (void)n_in; (void)out_size; (void)ws_size;
  const float* x     = (const float*)d_in[0];
  const float* fc_w  = (const float*)d_in[1];
  const float* fc_b  = (const float*)d_in[2];
  const float* wl[4]  = {(const float*)d_in[3],  (const float*)d_in[7],
                         (const float*)d_in[11], (const float*)d_in[15]};
  const float* wr[4]  = {(const float*)d_in[4],  (const float*)d_in[8],
                         (const float*)d_in[12], (const float*)d_in[16]};
  const float* att[4] = {(const float*)d_in[5],  (const float*)d_in[9],
                         (const float*)d_in[13], (const float*)d_in[17]};
  const float* bia[4] = {(const float*)d_in[6],  (const float*)d_in[10],
                         (const float*)d_in[14], (const float*)d_in[18]};
  const int* edge_index = (const int*)d_in[19];
  float* out = (float*)d_out;

  const int Cc[4] = {128, 128, 100, 100};

  // workspace carve-out (256B aligned slices)
  size_t off = 0;
  auto take = [&](size_t bytes) -> void* {
    void* p = (char*)d_ws + off;
    off += (bytes + 255) & ~(size_t)255;
    return p;
  };
  bf16_t* hidden = (bf16_t*)take((size_t)M_PAD * 256 * sizeof(bf16_t)); // padded rows
  float *xl[4], *xr[4];
  for (int i = 0; i < 4; ++i) {
    xl[i] = (float*)take((size_t)N_NODES * HEADS * Cc[i] * sizeof(float));
    xr[i] = (float*)take((size_t)N_NODES * HEADS * Cc[i] * sizeof(float));
  }
  float* logits = (float*)take((size_t)EPRIME * HEADS * sizeof(float));
  float* mbuf   = (float*)take((size_t)N_NODES * HEADS * sizeof(float));
  float* denom  = (float*)take((size_t)N_NODES * HEADS * sizeof(float));
  float* accum  = (float*)take((size_t)N_NODES * 512 * sizeof(float));

  // 1) hidden = relu(x @ fc_w + fc_b) -> bf16 [10000(+pad),256]
  gemm_wmma_kernel<0><<<dim3(cdiv_i(N_NODES, 128), cdiv_i(256, 64)), 256, 0, stream>>>(
      (const void*)x, fc_w, fc_b, (void*)hidden, N_NODES, 256, 4096);

  // 2) x_l / x_r projections for all four convs (A = bf16 hidden, rows padded)
  for (int i = 0; i < 4; ++i) {
    int HC = HEADS * Cc[i];
    dim3 g(cdiv_i(N_NODES, 128), cdiv_i(HC, 64));
    gemm_wmma_kernel<1><<<g, 256, 0, stream>>>(
        (const void*)hidden, wl[i], nullptr, (void*)xl[i], N_NODES, HC, 256);
    gemm_wmma_kernel<1><<<g, 256, 0, stream>>>(
        (const void*)hidden, wr[i], nullptr, (void*)xr[i], N_NODES, HC, 256);
  }

  // 3) per-conv attention softmax + scatter + finalize
  const long long ehThreads   = (long long)EPRIME * HEADS;        // 680K
  const long long waveThreads = ehThreads * 32;                   // 21.76M
  for (int i = 0; i < 4; ++i) {
    int C = Cc[i];
    int HC = HEADS * C;
    fill_u32_kernel<<<cdiv_i((long long)N_NODES * HEADS, 256), 256, 0, stream>>>(
        (unsigned int*)mbuf, 0xFF800000u /* -inf */, (long long)N_NODES * HEADS);
    fill_u32_kernel<<<cdiv_i((long long)N_NODES * HEADS, 256), 256, 0, stream>>>(
        (unsigned int*)denom, 0u, (long long)N_NODES * HEADS);
    fill_u32_kernel<<<cdiv_i((long long)N_NODES * HC, 256), 256, 0, stream>>>(
        (unsigned int*)accum, 0u, (long long)N_NODES * HC);

    edge_logits_max_kernel<<<cdiv_i(waveThreads, 256), 256, 0, stream>>>(
        xl[i], xr[i], att[i], edge_index, logits, mbuf, C);
    edge_softmax_kernel<<<cdiv_i(ehThreads, 256), 256, 0, stream>>>(
        edge_index, logits, mbuf, denom);
    edge_scatter_kernel<<<cdiv_i(waveThreads, 256), 256, 0, stream>>>(
        xl[i], edge_index, logits, denom, accum, C);

    float* obase = out + ((i == 1 || i == 3) ? (size_t)N_NODES * OUT_COLS : 0);
    int coloff = (i >= 2) ? 128 : 0;
    finalize_kernel<<<cdiv_i((long long)N_NODES * C, 256), 256, 0, stream>>>(
        accum, bia[i], obase, C, coloff);
  }
}
